// MambaBlock_26388279066976
// MI455X (gfx1250) — matrix operand order repaired and checked by reading, not compiled, not run on hardware
//
#include <hip/hip_runtime.h>
#include <hip/hip_bf16.h>
#include <math.h>

typedef __bf16 bf16_t;
typedef __attribute__((ext_vector_type(16))) __bf16 v16bf;
typedef __attribute__((ext_vector_type(8)))  float  v8f;

#define D_MODEL 1024
#define D_STATE 16
#define D_CONV  4
#define D_INNER 2048
#define DT_RANK 64
#define BATCH   2
#define SEQ_LEN 2048
#define NTOK    (BATCH * SEQ_LEN)       // 4096
#define XP_COLS 96                      // dt(64) + B(16) + C(16)
#define XP_LD   128                     // padded to a multiple of 64 for the GEMM
#define CHUNK   128
#define NCHUNK  (SEQ_LEN / CHUNK)       // 16
#define BK      32                      // GEMM k-step (WMMA depth)
#define BROW    40                      // LDS B-tile row stride (padded, bank-conflict-free)

// ---------------------------------------------------------------------------
__device__ __forceinline__ float silu_f(float z) {
    return z / (1.0f + __expf(-z));
}

__device__ __forceinline__ void async_b128_to_lds(unsigned lds_off, const void* gaddr) {
    // gfx1250 async copy: memory -> LDS directly, tracked by ASYNCcnt
    asm volatile("global_load_async_to_lds_b128 %0, %1, off"
                 :: "v"(lds_off), "v"(gaddr) : "memory");
}
__device__ __forceinline__ void wait_async0() {
    asm volatile("s_wait_asynccnt 0" ::: "memory");
}

// ---------------------------------------------------------------------------
// f32 -> bf16 conversion, with zero-padding beyond nvalid (for padded weights)
// ---------------------------------------------------------------------------
__global__ void k_f32_to_bf16(const float* __restrict__ s, bf16_t* __restrict__ d,
                              int nvalid, int ntotal) {
    int i = blockIdx.x * blockDim.x + threadIdx.x;
    if (i < ntotal) d[i] = (i < nvalid) ? (bf16_t)s[i] : (bf16_t)0.f;
}

// ---------------------------------------------------------------------------
// LayerNorm over D_MODEL=1024, one block per token, bf16 output (GEMM A matrix)
// ---------------------------------------------------------------------------
__global__ __launch_bounds__(256) void k_layernorm(const float* __restrict__ x,
                                                   const float* __restrict__ g,
                                                   const float* __restrict__ b,
                                                   bf16_t* __restrict__ xn) {
    __shared__ float s1[256], s2[256];
    const int m   = blockIdx.x;
    const int tid = threadIdx.x;
    const float* xr = x + (size_t)m * D_MODEL;
    float v[4], sum = 0.f, sq = 0.f;
#pragma unroll
    for (int j = 0; j < 4; ++j) {
        v[j] = xr[tid + 256 * j];
        sum += v[j]; sq += v[j] * v[j];
    }
    s1[tid] = sum; s2[tid] = sq;
    __syncthreads();
    for (int st = 128; st > 0; st >>= 1) {
        if (tid < st) { s1[tid] += s1[tid + st]; s2[tid] += s2[tid + st]; }
        __syncthreads();
    }
    const float mu   = s1[0] * (1.0f / D_MODEL);
    const float var  = s2[0] * (1.0f / D_MODEL) - mu * mu;
    const float rstd = rsqrtf(var + 1e-5f);
#pragma unroll
    for (int j = 0; j < 4; ++j) {
        int i = tid + 256 * j;
        xn[(size_t)m * D_MODEL + i] = (bf16_t)((v[j] - mu) * rstd * g[i] + b[i]);
    }
}

// ---------------------------------------------------------------------------
// bf16 WMMA GEMM:  C[M,N] = A[M,K] * W[N,K]^T   (f32 accumulate)
//   Requirements: M % 256 == 0, N % 64 == 0, K % 32 == 0 (caller pads).
//   block = 256 threads = 8 waves; wave -> 32 rows x 64 cols; block -> 256x64.
//   B strip (64 x 32) double-buffered in LDS via global_load_async_to_lds_b128,
//   shared by all 8 waves; A fragments loaded straight to VGPRs.
//   EPI 0: plain f32 store
//   EPI 2: softplus(acc + bias[n])          (dt_proj -> delta)
//   EPI 3: acc + resid[m*ldr + n]           (out_proj + residual)
// ---------------------------------------------------------------------------
template <int EPI>
__global__ __launch_bounds__(256) void k_gemm_bf16(const bf16_t* __restrict__ A,
                                                   const bf16_t* __restrict__ W,
                                                   float* __restrict__ C,
                                                   int K, int lda, int ldw, int ldc,
                                                   const float* __restrict__ bias,
                                                   const float* __restrict__ resid, int ldr) {
    __shared__ bf16_t sB[2][64 * BROW];       // two 64x32 bf16 tiles, padded rows
    const int tid  = threadIdx.x;
    const int lane = tid & 31;
    const int wave = tid >> 5;
    const int half = lane >> 4;               // K half-block selector (wave32 WMMA layout)
    const int l16  = lane & 15;
    const int row0 = blockIdx.y * 256 + wave * 32;
    const int col0 = blockIdx.x * 64;

    // async B loader: 64 rows x 32 k = 4KB per tile = 256 threads x one b128
    const int br = tid >> 2;                  // B row (0..63)
    const int bk = (tid & 3) * 8;             // k segment (0,8,16,24)
    const bf16_t* Wp = W + (size_t)(col0 + br) * ldw + bk;
    const unsigned lds0 = (unsigned)(size_t)&sB[0][br * BROW + bk];
    const unsigned lds1 = (unsigned)(size_t)&sB[1][br * BROW + bk];

    async_b128_to_lds(lds0, Wp);              // k0 = 0 tile
    wait_async0();
    __syncthreads();

    v8f acc[2][4] = {{v8f{}, v8f{}, v8f{}, v8f{}}, {v8f{}, v8f{}, v8f{}, v8f{}}};
    const bf16_t* Ar0 = A + (size_t)(row0 + l16) * lda;
    const bf16_t* Ar1 = A + (size_t)(row0 + 16 + l16) * lda;

    int buf = 0;
    for (int k0 = 0; k0 < K; k0 += BK) {
        const bool has_next = (k0 + BK < K);  // uniform
        if (has_next)
            async_b128_to_lds(buf ? lds0 : lds1, Wp + k0 + BK);  // prefetch next tile

        // A fragments (16x32): half 0 holds K [k0,k0+8)+[k0+16,k0+24), half 1 shifted by 8
        v16bf a0, a1;
        *((uint4*)&a0)     = *(const uint4*)(Ar0 + k0 + half * 8);
        *((uint4*)&a0 + 1) = *(const uint4*)(Ar0 + k0 + 16 + half * 8);
        *((uint4*)&a1)     = *(const uint4*)(Ar1 + k0 + half * 8);
        *((uint4*)&a1 + 1) = *(const uint4*)(Ar1 + k0 + 16 + half * 8);

        // all four B fragments first, so DS waits stagger instead of serializing
        v16bf bfr[4];
#pragma unroll
        for (int j = 0; j < 4; ++j) {
            const bf16_t* bp = &sB[buf][(j * 16 + l16) * BROW + half * 8];
            *((uint4*)&bfr[j])     = *(const uint4*)(bp);
            *((uint4*)&bfr[j] + 1) = *(const uint4*)(bp + 16);
        }
#pragma unroll
        for (int j = 0; j < 4; ++j)
            acc[0][j] = __builtin_amdgcn_wmma_f32_16x16x32_bf16(
                false, a0, false, bfr[j], (short)0, acc[0][j], false, false);
#pragma unroll
        for (int j = 0; j < 4; ++j)
            acc[1][j] = __builtin_amdgcn_wmma_f32_16x16x32_bf16(
                false, a1, false, bfr[j], (short)0, acc[1][j], false, false);

        if (has_next) {
            wait_async0();                    // own wave's async write landed
            __syncthreads();                  // everyone's writes landed
            buf ^= 1;
        }
    }

    // epilogue: one address per (i,j) sub-tile, then pointer-bump per row
#pragma unroll
    for (int i = 0; i < 2; ++i) {
        const int rbase = row0 + i * 16 + half * 8;   // D: VGPR r -> M = r + 8*half
#pragma unroll
        for (int j = 0; j < 4; ++j) {
            const int cn = col0 + j * 16 + l16;
            float* cp = C + (size_t)rbase * ldc + cn;
            float bv = 0.f;
            const float* rp = nullptr;
            if constexpr (EPI == 2) bv = bias[cn];
            if constexpr (EPI == 3) rp = resid + (size_t)rbase * ldr + cn;
#pragma unroll
            for (int r = 0; r < 8; ++r) {
                float v = acc[i][j][r];
                if constexpr (EPI == 2) {
                    float t = v + bv;
                    v = (t > 20.f) ? t : log1pf(__expf(t));
                }
                if constexpr (EPI == 3) {
                    v += *rp;
                    rp += ldr;
                }
                *cp = v;
                cp += ldc;
            }
        }
    }
}

// ---------------------------------------------------------------------------
// Causal depthwise conv1d (K=4) + bias + SiLU. Reads x-half of xz (ld 4096).
// Writes f32 (scan input) and bf16 (x_proj GEMM A matrix).
// ---------------------------------------------------------------------------
__global__ void k_conv(const float* __restrict__ xz, const float* __restrict__ cw,
                       const float* __restrict__ cb, float* __restrict__ xcf,
                       bf16_t* __restrict__ xcb) {
    const int idx = blockIdx.x * blockDim.x + threadIdx.x;   // NTOK*D_INNER
    if (idx >= NTOK * D_INNER) return;
    const int d = idx & (D_INNER - 1);
    const int m = idx >> 11;            // token index (b*L + l)
    const int l = m & (SEQ_LEN - 1);
    float acc = cb[d];
#pragma unroll
    for (int k = 0; k < D_CONV; ++k) {
        const int ls = l + k - (D_CONV - 1);
        if (ls >= 0)
            acc += xz[(size_t)(m + k - (D_CONV - 1)) * (2 * D_INNER) + d] * cw[d * D_CONV + k];
    }
    const float s = silu_f(acc);
    xcf[idx] = s;
    xcb[idx] = (bf16_t)s;
}

// ---------------------------------------------------------------------------
// Extract dt columns (0..63) of padded xp into a compact bf16 matrix
// ---------------------------------------------------------------------------
__global__ void k_extract_dt(const float* __restrict__ xp, bf16_t* __restrict__ dt) {
    const int i = blockIdx.x * blockDim.x + threadIdx.x;   // NTOK*DT_RANK
    if (i >= NTOK * DT_RANK) return;
    const int m = i >> 6, j = i & 63;
    dt[i] = (bf16_t)xp[(size_t)m * XP_LD + j];
}

// ---------------------------------------------------------------------------
// Chunked selective scan.
//  PASS 0: per-chunk partial:  h_part, aprod = prod(dA)   -> state buffer
//  PASS 1: replay chunk from injected initial h, emit y = (h.C + x*Dp)*silu(z)
// Grid: (D_INNER/256, NCHUNK, BATCH); lane = one channel d; B/C staged in LDS.
// state layout: [(b*D_INNER+d)*NCHUNK + chunk] * 32 floats  (h[16] | aprod/init[16])
// ---------------------------------------------------------------------------
template <int PASS>
__global__ __launch_bounds__(256) void k_scan(const float* __restrict__ xcf,
                                              const float* __restrict__ delta,
                                              const float* __restrict__ xp,
                                              const float* __restrict__ logA,
                                              const float* __restrict__ Dp,
                                              const float* __restrict__ xz,
                                              float* __restrict__ state,
                                              bf16_t* __restrict__ ybf) {
    __shared__ float sB[CHUNK][D_STATE];
    __shared__ float sC[CHUNK][D_STATE];
    const int d     = blockIdx.x * 256 + threadIdx.x;
    const int chunk = blockIdx.y;
    const int b     = blockIdx.z;
    const int t0    = chunk * CHUNK;

    for (int i = threadIdx.x; i < CHUNK * D_STATE; i += 256) {
        const int t = i >> 4, s = i & 15;
        const float* row = xp + (size_t)(b * SEQ_LEN + t0 + t) * XP_LD;
        sB[t][s] = row[DT_RANK + s];
        sC[t][s] = row[DT_RANK + D_STATE + s];
    }
    __syncthreads();

    float As[D_STATE], h[D_STATE], ap[D_STATE];
#pragma unroll
    for (int s = 0; s < D_STATE; ++s) As[s] = -__expf(logA[d * D_STATE + s]);

    const size_t sbase = ((size_t)(b * D_INNER + d) * NCHUNK + chunk) * 32;
    if (PASS == 0) {
#pragma unroll
        for (int s = 0; s < D_STATE; ++s) { h[s] = 0.f; ap[s] = 1.f; }
    } else {
#pragma unroll
        for (int s = 0; s < D_STATE; ++s) h[s] = state[sbase + 16 + s];  // injected init
    }
    const float Dpd = Dp[d];

    for (int t = 0; t < CHUNK; ++t) {
        const size_t m  = (size_t)(b * SEQ_LEN + t0 + t);
        const float  xt = xcf[m * D_INNER + d];
        const float  dl = delta[m * D_INNER + d];
        const float  dbx = dl * xt;
        float y = 0.f;
#pragma unroll
        for (int s = 0; s < D_STATE; ++s) {
            const float dA = __expf(dl * As[s]);
            h[s] = dA * h[s] + dbx * sB[t][s];
            if (PASS == 0) ap[s] *= dA;
            else           y += h[s] * sC[t][s];
        }
        if (PASS == 1) {
            const float z = xz[m * (2 * D_INNER) + D_INNER + d];
            ybf[m * D_INNER + d] = (bf16_t)((y + xt * Dpd) * silu_f(z));
        }
    }
    if (PASS == 0) {
#pragma unroll
        for (int s = 0; s < D_STATE; ++s) {
            state[sbase + s]      = h[s];
            state[sbase + 16 + s] = ap[s];
        }
    }
}

// Sequential combine across the 16 chunks (cheap; 4096 independent lanes).
// Rewrites slot [16..31] of each chunk with the initial state for that chunk.
__global__ void k_scan_combine(float* __restrict__ state) {
    const int i = blockIdx.x * 256 + threadIdx.x;   // b*D_INNER + d
    if (i >= BATCH * D_INNER) return;
    float carry[D_STATE];
#pragma unroll
    for (int s = 0; s < D_STATE; ++s) carry[s] = 0.f;
    for (int c = 0; c < NCHUNK; ++c) {
        const size_t sbase = ((size_t)i * NCHUNK + c) * 32;
#pragma unroll
        for (int s = 0; s < D_STATE; ++s) {
            const float hh = state[sbase + s];
            const float aa = state[sbase + 16 + s];
            state[sbase + 16 + s] = carry[s];          // init for chunk c
            carry[s] = aa * carry[s] + hh;
        }
    }
}

// ---------------------------------------------------------------------------
// workspace layout
// ---------------------------------------------------------------------------
static constexpr size_t AL(size_t x) { return (x + 255) & ~(size_t)255; }
static constexpr size_t OFF_XN    = 0;
static constexpr size_t OFF_WIN   = OFF_XN    + AL((size_t)NTOK * D_MODEL * 2);
static constexpr size_t OFF_XZ    = OFF_WIN   + AL((size_t)2 * D_INNER * D_MODEL * 2);
static constexpr size_t OFF_XCF   = OFF_XZ    + AL((size_t)NTOK * 2 * D_INNER * 4);
static constexpr size_t OFF_XCB   = OFF_XCF   + AL((size_t)NTOK * D_INNER * 4);
static constexpr size_t OFF_WXP   = OFF_XCB   + AL((size_t)NTOK * D_INNER * 2);
static constexpr size_t OFF_XP    = OFF_WXP   + AL((size_t)XP_LD * D_INNER * 2);
static constexpr size_t OFF_DTB   = OFF_XP    + AL((size_t)NTOK * XP_LD * 4);
static constexpr size_t OFF_WDT   = OFF_DTB   + AL((size_t)NTOK * DT_RANK * 2);
static constexpr size_t OFF_DELTA = OFF_WDT   + AL((size_t)D_INNER * DT_RANK * 2);
static constexpr size_t OFF_STATE = OFF_DELTA + AL((size_t)NTOK * D_INNER * 4);
static constexpr size_t OFF_YBF   = OFF_STATE + AL((size_t)BATCH * D_INNER * NCHUNK * 32 * 4);
static constexpr size_t OFF_WOUT  = OFF_YBF   + AL((size_t)NTOK * D_INNER * 2);

extern "C" void kernel_launch(void* const* d_in, const int* in_sizes, int n_in,
                              void* d_out, int out_size, void* d_ws, size_t ws_size,
                              hipStream_t stream) {
    const float* x         = (const float*)d_in[0];
    const float* norm_g    = (const float*)d_in[1];
    const float* norm_b    = (const float*)d_in[2];
    const float* in_proj_w = (const float*)d_in[3];
    const float* conv_w    = (const float*)d_in[4];
    const float* conv_b    = (const float*)d_in[5];
    const float* x_proj_w  = (const float*)d_in[6];
    const float* dt_proj_w = (const float*)d_in[7];
    const float* dt_proj_b = (const float*)d_in[8];
    const float* log_A     = (const float*)d_in[9];
    const float* Dp        = (const float*)d_in[10];
    const float* out_proj_w= (const float*)d_in[11];
    float* out = (float*)d_out;

    char* ws = (char*)d_ws;
    bf16_t* xn_bf   = (bf16_t*)(ws + OFF_XN);
    bf16_t* win_bf  = (bf16_t*)(ws + OFF_WIN);
    float*  xz      = (float*) (ws + OFF_XZ);
    float*  xcf     = (float*) (ws + OFF_XCF);
    bf16_t* xcb     = (bf16_t*)(ws + OFF_XCB);
    bf16_t* wxp_bf  = (bf16_t*)(ws + OFF_WXP);
    float*  xp      = (float*) (ws + OFF_XP);
    bf16_t* dt_bf   = (bf16_t*)(ws + OFF_DTB);
    bf16_t* wdt_bf  = (bf16_t*)(ws + OFF_WDT);
    float*  delta   = (float*) (ws + OFF_DELTA);
    float*  state   = (float*) (ws + OFF_STATE);
    bf16_t* y_bf    = (bf16_t*)(ws + OFF_YBF);
    bf16_t* wout_bf = (bf16_t*)(ws + OFF_WOUT);

    // 1. weight downconversion to bf16 (x_proj_w zero-padded 96 -> 128 rows)
    {
        int n;
        n = 2 * D_INNER * D_MODEL;
        k_f32_to_bf16<<<(n + 255) / 256, 256, 0, stream>>>(in_proj_w, win_bf, n, n);
        int nv = XP_COLS * D_INNER, nt = XP_LD * D_INNER;
        k_f32_to_bf16<<<(nt + 255) / 256, 256, 0, stream>>>(x_proj_w, wxp_bf, nv, nt);
        n = D_INNER * DT_RANK;
        k_f32_to_bf16<<<(n + 255) / 256, 256, 0, stream>>>(dt_proj_w, wdt_bf, n, n);
        n = D_MODEL * D_INNER;
        k_f32_to_bf16<<<(n + 255) / 256, 256, 0, stream>>>(out_proj_w, wout_bf, n, n);
    }

    // 2. LayerNorm -> bf16 activations
    k_layernorm<<<NTOK, 256, 0, stream>>>(x, norm_g, norm_b, xn_bf);

    // 3. in_proj GEMM: xz[4096,4096] = xn[4096,1024] * in_proj_w[4096,1024]^T
    k_gemm_bf16<0><<<dim3(2 * D_INNER / 64, NTOK / 256), 256, 0, stream>>>(
        xn_bf, win_bf, xz, D_MODEL,
        D_MODEL, D_MODEL, 2 * D_INNER, nullptr, nullptr, 0);

    // 4. causal depthwise conv + SiLU
    k_conv<<<(NTOK * D_INNER + 255) / 256, 256, 0, stream>>>(xz, conv_w, conv_b, xcf, xcb);

    // 5. x_proj GEMM: xp[4096,128p] = xconv[4096,2048] * x_proj_w[128p,2048]^T
    k_gemm_bf16<0><<<dim3(XP_LD / 64, NTOK / 256), 256, 0, stream>>>(
        xcb, wxp_bf, xp, D_INNER,
        D_INNER, D_INNER, XP_LD, nullptr, nullptr, 0);

    // 6. dt extraction + dt_proj GEMM with fused bias+softplus -> delta
    k_extract_dt<<<(NTOK * DT_RANK + 255) / 256, 256, 0, stream>>>(xp, dt_bf);
    k_gemm_bf16<2><<<dim3(D_INNER / 64, NTOK / 256), 256, 0, stream>>>(
        dt_bf, wdt_bf, delta, DT_RANK,
        DT_RANK, DT_RANK, D_INNER, dt_proj_b, nullptr, 0);

    // 7. chunked selective scan (3 passes) fused with D-skip + SiLU(z) gating
    dim3 sg(D_INNER / 256, NCHUNK, BATCH);
    k_scan<0><<<sg, 256, 0, stream>>>(xcf, delta, xp, log_A, Dp, xz, state, y_bf);
    k_scan_combine<<<(BATCH * D_INNER + 255) / 256, 256, 0, stream>>>(state);
    k_scan<1><<<sg, 256, 0, stream>>>(xcf, delta, xp, log_A, Dp, xz, state, y_bf);

    // 8. out_proj GEMM with fused residual add -> d_out
    k_gemm_bf16<3><<<dim3(D_MODEL / 64, NTOK / 256), 256, 0, stream>>>(
        y_bf, wout_bf, out, D_INNER,
        D_INNER, D_INNER, D_MODEL, nullptr, x, D_MODEL);

    (void)in_sizes; (void)n_in; (void)out_size; (void)ws_size;
}
